// FullAttention_13383118094711
// MI455X (gfx1250) — matrix operand order repaired and checked by reading, not compiled, not run on hardware
//
#include <hip/hip_runtime.h>

typedef __attribute__((ext_vector_type(16))) _Float16 v16h;
typedef __attribute__((ext_vector_type(2)))  __fp16   v2hf;  // cvt_pkrtz return type
typedef __attribute__((ext_vector_type(8)))  float    v8f;

namespace {
constexpr int Bdim = 2, Ldim = 4096, Hdim = 8, Edim = 64, Sdim = 4096, Ddim = 64;
constexpr int NW  = 8;    // waves per block
constexpr int BM  = 128;  // query rows per block: 8 waves x 16
constexpr int BN  = 64;   // keys per chunk (4 n-tiles, 2 WMMA k-chunks for P*V)
// 72 halves = 144 B stride: 16B-aligned rows (144 = 9*16) enabling ds_load_b128
// operand fetches; 144/16 = 9 odd -> 16 lanes cover 16 distinct 16B bank groups.
constexpr int KST = 72;
constexpr int VST = 72;
constexpr int PST = 72;
}

union U16 { v16h v; _Float16 h[16]; unsigned u[8]; uint4 q[2]; };
union U2  { v2hf v; unsigned u; };

__device__ inline float hmax16(float x) {
  x = fmaxf(x, __shfl_xor(x, 1));
  x = fmaxf(x, __shfl_xor(x, 2));
  x = fmaxf(x, __shfl_xor(x, 4));
  x = fmaxf(x, __shfl_xor(x, 8));
  return x;
}

__global__ __launch_bounds__(256)
void fa_causal_wmma(const float* __restrict__ Q, const float* __restrict__ K,
                    const float* __restrict__ V, float* __restrict__ O) {
  __shared__ __align__(16) _Float16 sK[2][BN * KST];      // double-buffered K chunk, [key][e]
  __shared__ __align__(16) _Float16 sV[2][Ddim * VST];    // double-buffered V^T chunk, [d][key]
  __shared__ __align__(16) _Float16 sP[NW * 16 * PST];    // per-wave P scratch, [row][key]

  const int tid  = threadIdx.x;
  const int wave = tid >> 5;
  const int lane = tid & 31;
  const int l15  = lane & 15;
  const int hl   = (lane >> 4) & 1;

  const int qb    = blockIdx.x * BM;
  const int head  = blockIdx.y;
  const int batch = blockIdx.z;

  // ---- Q tile (16 rows per wave) into WMMA A-layout regs, pre-scaled ----
  const float scale = 0.125f;  // 1/sqrt(64)
  const int qrowA = qb + wave * 16 + l15;  // A-layout: lane&15 = row
  const float* Qrow = Q + (((size_t)batch * Ldim + qrowA) * Hdim + head) * Edim;
  U16 qa[2];
  #pragma unroll
  for (int g = 0; g < 2; ++g) {
    #pragma unroll
    for (int h2 = 0; h2 < 16; ++h2) {
      const int e = 32 * g + 16 * hl + h2;
      qa[g].h[h2] = (_Float16)(scale * Qrow[e]);
    }
  }

  // Constant all-ones B operand: P @ ones accumulates the softmax denominator
  // on the matrix unit, replicated across all lanes of each row.
  U16 onesB;
  #pragma unroll
  for (int h2 = 0; h2 < 16; ++h2) onesB.h[h2] = (_Float16)1.0f;

  v8f acc[5];  // [0..3]: output d-tiles, [4]: running softmax denominator
  #pragma unroll
  for (int c = 0; c < 5; ++c) acc[c] = {};
  float rmax[8];
  #pragma unroll
  for (int i = 0; i < 8; ++i) rmax[i] = -1e30f;

  const int nEnd = qb + BM;          // causal bound for this block
  const int qmin = qb + wave * 16;   // first row owned by this wave
  const int qmax = qmin + 15;        // last row owned by this wave
  const float* Kbh = K + (size_t)batch * Sdim * Hdim * Edim + (size_t)head * Edim;
  const float* Vbh = V + (size_t)batch * Sdim * Hdim * Ddim + (size_t)head * Ddim;
  _Float16* pw = &sP[wave * 16 * PST];

  // Per-thread staging slots: 4 float4 units of K and V each (64 rows x 16 units / 256 thr)
  const int srow = tid >> 4;          // rows srow, srow+16, srow+32, srow+48
  const int se4  = (tid & 15) * 4;
  float4 kq[4], vq[4];
  #pragma unroll
  for (int it = 0; it < 4; ++it) {    // prime the pipeline with chunk 0
    const int row = srow + it * 16;
    kq[it] = *(const float4*)(Kbh + (size_t)row * Hdim * Edim + se4);
    vq[it] = *(const float4*)(Vbh + (size_t)row * Hdim * Ddim + se4);
  }

  int buf = 0;
  for (int n0 = 0; n0 < nEnd; n0 += BN, buf ^= 1) {
    // ---- commit staged registers (f32 -> f16, packed) into LDS buffer `buf` ----
    #pragma unroll
    for (int it = 0; it < 4; ++it) {
      const int row = srow + it * 16;
      U2 klo, khi;
      klo.v = __builtin_amdgcn_cvt_pkrtz(kq[it].x, kq[it].y);
      khi.v = __builtin_amdgcn_cvt_pkrtz(kq[it].z, kq[it].w);
      *(uint2*)&sK[buf][row * KST + se4] = uint2{klo.u, khi.u};  // 8B-aligned
      _Float16* dv = &sV[buf][0];
      dv[(se4 + 0) * VST + row] = (_Float16)vq[it].x;
      dv[(se4 + 1) * VST + row] = (_Float16)vq[it].y;
      dv[(se4 + 2) * VST + row] = (_Float16)vq[it].z;
      dv[(se4 + 3) * VST + row] = (_Float16)vq[it].w;
    }
    // ---- issue next chunk's global loads; latency hidden behind barrier+compute ----
    if (n0 + BN < nEnd) {
      #pragma unroll
      for (int it = 0; it < 4; ++it) {
        const int row = n0 + BN + srow + it * 16;
        kq[it] = *(const float4*)(Kbh + (size_t)row * Hdim * Edim + se4);
        vq[it] = *(const float4*)(Vbh + (size_t)row * Hdim * Ddim + se4);
      }
    }
    __syncthreads();  // buffer `buf` visible to all waves

    // Wave-uniform causal guard (scalar branch; EXEC stays all-ones for WMMA)
    const int active   = __builtin_amdgcn_readfirstlane((n0 <= qmax) ? 1 : 0);
    const int needMask = __builtin_amdgcn_readfirstlane((n0 + BN - 1 > qmin) ? 1 : 0);

    if (active) {
      const _Float16* cK = &sK[buf][0];
      const _Float16* cV = &sV[buf][0];

      // ---- scores: 4 n-tiles in pairs; B operands are 2 x b128 per tile ----
      v8f sc[4];
      #pragma unroll
      for (int tp = 0; tp < 2; ++tp) {
        const int t0 = 2 * tp, t1 = t0 + 1;
        // B-layout: lane&15 = column (key); halves = 16 consecutive e at 32g+16hl
        const _Float16* r0 = &cK[(16 * t0 + l15) * KST + 16 * hl];
        const _Float16* r1 = &cK[(16 * t1 + l15) * KST + 16 * hl];
        U16 a0, b0, a1, b1;
        a0.q[0] = *(const uint4*)(r0);      a0.q[1] = *(const uint4*)(r0 + 8);
        b0.q[0] = *(const uint4*)(r0 + 32); b0.q[1] = *(const uint4*)(r0 + 40);
        a1.q[0] = *(const uint4*)(r1);      a1.q[1] = *(const uint4*)(r1 + 8);
        b1.q[0] = *(const uint4*)(r1 + 32); b1.q[1] = *(const uint4*)(r1 + 40);
        v8f z = {};
        v8f w0 = __builtin_amdgcn_wmma_f32_16x16x32_f16(false, qa[0].v, false, a0.v,
                                                        (short)0, z, false, false);
        v8f w1 = __builtin_amdgcn_wmma_f32_16x16x32_f16(false, qa[0].v, false, a1.v,
                                                        (short)0, z, false, false);
        sc[t0] = __builtin_amdgcn_wmma_f32_16x16x32_f16(false, qa[1].v, false, b0.v,
                                                        (short)0, w0, false, false);
        sc[t1] = __builtin_amdgcn_wmma_f32_16x16x32_f16(false, qa[1].v, false, b1.v,
                                                        (short)0, w1, false, false);
      }

      // ---- online softmax (f32): row max via lane butterfly; denominator via WMMA ----
      #pragma unroll
      for (int i = 0; i < 8; ++i) {
        const int q = qmin + i + hl * 8;  // C-layout row for VGPR i
        float s[4];
        #pragma unroll
        for (int t = 0; t < 4; ++t) s[t] = sc[t][i];
        if (needMask) {
          #pragma unroll
          for (int t = 0; t < 4; ++t)
            if (n0 + 16 * t + l15 > q) s[t] = -1e30f;
        }
        const float cm   = hmax16(fmaxf(fmaxf(s[0], s[1]), fmaxf(s[2], s[3])));
        const float mnew = fmaxf(rmax[i], cm);
        const float f    = __expf(rmax[i] - mnew);
        rmax[i] = mnew;
        #pragma unroll
        for (int c2 = 0; c2 < 5; ++c2) acc[c2][i] *= f;  // rescale output AND denominator
        const int m = i + hl * 8;
        #pragma unroll
        for (int t = 0; t < 4; ++t)
          pw[m * PST + 16 * t + l15] = (_Float16)__expf(s[t] - mnew);
      }

      // P scratch is wave-private; same-wave LDS ops are processed in order, so a
      // dscnt wait + compiler barrier replaces a block-wide __syncthreads.
      asm volatile("s_wait_dscnt 0x0" ::: "memory");

      // ---- O += P @ V (and denom += P @ ones) per k-chunk ----
      // A-layout: 16 consecutive k at 32kc + 16hl (two 16B runs).
      U16 pa[2];
      #pragma unroll
      for (int kc = 0; kc < 2; ++kc) {
        const _Float16* pr = &pw[l15 * PST + 32 * kc + 16 * hl];
        pa[kc].q[0] = *(const uint4*)(pr);
        pa[kc].q[1] = *(const uint4*)(pr + 8);
      }
      #pragma unroll
      for (int kc = 0; kc < 2; ++kc) {
        U16 bv[4];
        #pragma unroll
        for (int c2 = 0; c2 < 4; ++c2) {
          // B-layout: 16 consecutive k at 32kc+16hl within row d = 16c2+l15
          const _Float16* vr = &cV[(16 * c2 + l15) * VST + 32 * kc + 16 * hl];
          bv[c2].q[0] = *(const uint4*)(vr);
          bv[c2].q[1] = *(const uint4*)(vr + 8);
        }
        #pragma unroll
        for (int c2 = 0; c2 < 4; ++c2)
          acc[c2] = __builtin_amdgcn_wmma_f32_16x16x32_f16(false, pa[kc].v, false, bv[c2].v,
                                                           (short)0, acc[c2], false, false);
        acc[4] = __builtin_amdgcn_wmma_f32_16x16x32_f16(false, pa[kc].v, false, onesB.v,
                                                        (short)0, acc[4], false, false);
      }
    }
  }

  // ---- epilogue: O = acc / denominator (acc[4], lane-replicated) ----
  #pragma unroll
  for (int i = 0; i < 8; ++i) {
    const float inv = 1.f / acc[4][i];
    const int q = qmin + i + hl * 8;
    float* orow = O + (((size_t)batch * Ldim + q) * Hdim + head) * Ddim;
    #pragma unroll
    for (int c2 = 0; c2 < 4; ++c2) orow[16 * c2 + l15] = acc[c2][i] * inv;
  }
}

extern "C" void kernel_launch(void* const* d_in, const int* in_sizes, int n_in,
                              void* d_out, int out_size, void* d_ws, size_t ws_size,
                              hipStream_t stream) {
  (void)in_sizes; (void)n_in; (void)out_size; (void)d_ws; (void)ws_size;
  const float* Q = (const float*)d_in[0];
  const float* K = (const float*)d_in[1];
  const float* V = (const float*)d_in[2];
  float* O = (float*)d_out;
  dim3 grid(Ldim / BM, Hdim, Bdim);  // 32 x 8 x 2 = 512 workgroups
  dim3 block(256);                   // 8 waves (wave32)
  fa_causal_wmma<<<grid, block, 0, stream>>>(Q, K, V, O);
}
